// SecondOrderRelationalAttention_41798621724803
// MI455X (gfx1250) — matrix operand order repaired and checked by reading, not compiled, not run on hardware
//
#include <hip/hip_runtime.h>
#include <hip/hip_bf16.h>

typedef _Float16 half_t;
typedef __attribute__((ext_vector_type(16))) _Float16 v16h;
typedef __attribute__((ext_vector_type(8)))  _Float16 v8h;
typedef __attribute__((ext_vector_type(8)))  float    v8f;
typedef __attribute__((ext_vector_type(4)))  float    v4f;
typedef int v4i __attribute__((vector_size(16)));   // matches builtin param type

#define BB 2
#define CC 1024
#define DDIM 512
#define HH 8
#define DH 64
#define CHUNK 64
#define NC (CC / CHUNK)

#if __has_builtin(__builtin_amdgcn_global_load_async_to_lds_b128)
#define HAVE_ASYNC_LDS 1
#endif

// ---------------------------------------------------------------------------
// Async global -> LDS copy of 16 bytes (per-lane), ASYNCcnt-tracked on CDNA5.
// Builtin signature (probe-revealed): (v4i as1* src, v4i as3* dst, Iimm, Iimm).
// Falls back to a VGPR round-trip copy when the builtin is unavailable.
// ---------------------------------------------------------------------------
static __device__ __forceinline__ void copy16_g2l(void* lds, const void* g) {
#ifdef HAVE_ASYNC_LDS
  __builtin_amdgcn_global_load_async_to_lds_b128(
      (__attribute__((address_space(1))) v4i*)(uintptr_t)g,
      (__attribute__((address_space(3))) v4i*)(uint32_t)(uintptr_t)lds, 0, 0);
#else
  *(v4f*)lds = *(const v4f*)g;
#endif
}
static __device__ __forceinline__ void wait_async_lds() {
#ifdef HAVE_ASYNC_LDS
#if __has_builtin(__builtin_amdgcn_s_wait_asynccnt)
  __builtin_amdgcn_s_wait_asynccnt(0);
#else
  asm volatile("s_wait_asynccnt 0x0" ::: "memory");
#endif
#endif
}

// ---------------------------------------------------------------------------
// Fragment helpers (layouts per CDNA5 ISA 7.12.2, wave32)
// A 16x32 f16 : lane L holds row L%16; halves [base0..base0+8) and
//               [base0+16..base0+24) with base0 = (L/16)*8
// B 32x16 f16 : lane L holds col L%16; same two K halves as A (base0 = (L/16)*8)
// C/D 16x16 f32: lane L holds col L%16; VGPR r -> row r + 8*(L/16)
// ---------------------------------------------------------------------------
static __device__ __forceinline__ v16h frag2(const half_t* p0, const half_t* p1) {
  union { v16h v; v8h h[2]; } u;
  u.h[0] = *(const v8h*)p0;
  u.h[1] = *(const v8h*)p1;
  return u.v;
}

static __device__ __forceinline__ v8f wmma_f16(v16h a, v16h b, v8f c) {
  return __builtin_amdgcn_wmma_f32_16x16x32_f16(false, a, false, b, (short)0, c,
                                                false, false);
}

static __device__ __forceinline__ float fast_exp(float x) {
#if __has_builtin(__builtin_amdgcn_exp2f)
  return __builtin_amdgcn_exp2f(x * 1.4426950408889634f);
#else
  return __expf(x);
#endif
}
static __device__ __forceinline__ float fast_tanh(float x) {
#if __has_builtin(__builtin_amdgcn_tanhf)
  return __builtin_amdgcn_tanhf(x);
#else
  return tanhf(x);
#endif
}
static __device__ __forceinline__ float gelu_f(float x) {
  float x3 = x * x * x;
  return 0.5f * x * (1.0f + fast_tanh(0.7978845608028654f * (x + 0.044715f * x3)));
}

// ---------------------------------------------------------------------------
// f32 -> f16 conversion
// ---------------------------------------------------------------------------
__global__ void f32_to_f16_kernel(const float* __restrict__ src,
                                  half_t* __restrict__ dst, int n) {
  int i = blockIdx.x * 256 + threadIdx.x;
  if (i < n) dst[i] = (half_t)src[i];
}

// split w1 (64 x 192) into w1_q / w1_k / w1_qk, each (64 x 64), f16
__global__ void split_w1_kernel(const float* __restrict__ w1,
                                half_t* __restrict__ wq, half_t* __restrict__ wk,
                                half_t* __restrict__ wqk) {
  int i = blockIdx.x * 256 + threadIdx.x;  // 0..4095
  int e = i >> 6, dd = i & 63;
  wq[i]  = (half_t)w1[e * 192 + dd];
  wk[i]  = (half_t)w1[e * 192 + 64 + dd];
  wqk[i] = (half_t)w1[e * 192 + 128 + dd];
}

// ---------------------------------------------------------------------------
// Generic WMMA GEMM: Y(MxN) = A(MxK,f16,rowmajor) @ W(NxK,f16,rowmajor)^T + bias
// mode 0: f32 output, row-major (M,N)
// mode 1: f16 output, remapped to (B,H,C,d) head layout  (for Q/K/V)
// One wave computes one 16x16 tile; 8 waves per block.
// ---------------------------------------------------------------------------
__global__ __launch_bounds__(256) void gemm_wmma_kernel(
    const half_t* __restrict__ A, const half_t* __restrict__ W,
    const float* __restrict__ bias, void* __restrict__ out, int M, int N, int K,
    int mode) {
  const int lane = threadIdx.x & 31;
  const int ln = lane & 15, lh = lane >> 4;
  const int tile = blockIdx.x * 8 + (threadIdx.x >> 5);
  const int tilesN = N >> 4;
  const int m0 = (tile / tilesN) * 16;
  const int n0 = (tile % tilesN) * 16;
  if (m0 >= M) return;

  v8f acc = {0.f, 0.f, 0.f, 0.f, 0.f, 0.f, 0.f, 0.f};
  const half_t* arow = A + (size_t)(m0 + ln) * K;
  const half_t* wrow = W + (size_t)(n0 + ln) * K;
  for (int k0 = 0; k0 < K; k0 += 32) {
    v16h a = frag2(arow + k0 + lh * 8, arow + k0 + 16 + lh * 8);
    v16h b = frag2(wrow + k0 + lh * 8, wrow + k0 + 16 + lh * 8);
    acc = wmma_f16(a, b, acc);
  }
  const float bv = bias ? bias[n0 + ln] : 0.0f;
  if (mode == 0) {
    float* o = (float*)out;
    for (int r = 0; r < 8; ++r) {
      int m = m0 + r + 8 * lh, nn = n0 + ln;
      o[(size_t)m * N + nn] = acc[r] + bv;
    }
  } else {
    half_t* o = (half_t*)out;
    for (int r = 0; r < 8; ++r) {
      int m = m0 + r + 8 * lh, nn = n0 + ln;
      int bb = m / CC, c = m % CC, h = nn / DH, dd = nn & (DH - 1);
      o[(((size_t)(bb * HH + h)) * CC + c) * DH + dd] = (half_t)(acc[r] + bv);
    }
  }
}

// ---------------------------------------------------------------------------
// Second-order relational attention, one workgroup per (b, h, query-chunk).
// 8 waves / 256 threads. Flash-style online softmax over 16 key blocks of 64,
// with async-LDS double-buffered K / KP / V staging.
// For each key j:   hid(64x64) = (q .* k[j]) @ w1_qk^T + qp + kp[j] + b1
//                   score(i,j) = (sum_e gelu(hid) * w2[e] + b2) / sqrt(d)
// Then O += softmax(P) @ V via WMMA.
// ---------------------------------------------------------------------------
__global__ __launch_bounds__(256, 1) void sora_attn_kernel(
    const half_t* __restrict__ qh, const half_t* __restrict__ kh,
    const half_t* __restrict__ vh, const float* __restrict__ qp,
    const float* __restrict__ kp, const half_t* __restrict__ w1qk,
    const float* __restrict__ b1, const float* __restrict__ w2,
    const float* __restrict__ b2, half_t* __restrict__ oh) {
  __shared__ half_t sQ[CHUNK * DH];        // q chunk (f16)              8 KB
  __shared__ half_t sW1[DH * DH];          // w1_qk (f16)                8 KB
  __shared__ half_t sK[2][CHUNK * DH];     // key blocks (dbl-buf)      16 KB
  __shared__ half_t sVraw[2][CHUNK * DH];  // V blocks raw (dbl-buf)    16 KB
  __shared__ half_t sVt[DH * CHUNK];       // V block transposed         8 KB
  __shared__ float  sKP[2][CHUNK * DH];    // kp blocks (dbl-buf)       32 KB
  __shared__ float  sS[CHUNK * CHUNK];     // scores                    16 KB
  __shared__ half_t sP[CHUNK * CHUNK];     // softmax probs (f16)        8 KB
  __shared__ float  sQPf[16 * 32 * 8];     // qp in D-fragment layout   16 KB
  __shared__ float  sMrow[CHUNK], sLrow[CHUNK], sScale[CHUNK];
  __shared__ float  sW2[DH], sB1[DH];

  const int tid = threadIdx.x;
  const int wave = tid >> 5;
  const int lane = tid & 31;
  const int ln = lane & 15;
  const int lh = lane >> 4;

  const int qc = blockIdx.x % NC;
  const int hh = (blockIdx.x / NC) % HH;
  const int b  = blockIdx.x / (NC * HH);
  const size_t base = (size_t)(b * HH + hh) * CC * DH;
  const int q0 = qc * CHUNK;

  // ---- initial staging: q chunk + w1_qk (async), qp fragments (manual) ----
  for (int idx = tid; idx < 512; idx += 256) {
    copy16_g2l(&sQ[idx * 8], qh + base + (size_t)q0 * DH + idx * 8);
    copy16_g2l(&sW1[idx * 8], w1qk + idx * 8);
  }
  if (tid < DH) { sW2[tid] = w2[tid]; sB1[tid] = b1[tid]; }
  if (tid < CHUNK) { sMrow[tid] = -3.0e38f; sLrow[tid] = 0.0f; sScale[tid] = 0.f; }
  for (int t = 0; t < 16; ++t) {  // qp -> C/D fragment layout [et*4+mt][lane][r]
    int idx = t * 256 + tid;
    int fid = idx >> 8;
    int rem = idx & 255;
    int l2 = rem >> 3, r = rem & 7;
    int et = fid >> 2, mt = fid & 3;
    int m = mt * 16 + r + 8 * (l2 >> 4);
    int e = et * 16 + (l2 & 15);
    sQPf[idx] = qp[base + (size_t)(q0 + m) * DH + e];
  }
  // ---- prefetch key block 0 into buffer 0 ----
  {
    const half_t* kg = kh + base;
    const half_t* vg = vh + base;
    const float*  pg = kp + base;
    for (int idx = tid; idx < 512; idx += 256) {
      copy16_g2l(&sK[0][idx * 8], kg + idx * 8);
      copy16_g2l(&sVraw[0][idx * 8], vg + idx * 8);
    }
    for (int idx = tid; idx < 1024; idx += 256)
      copy16_g2l(&sKP[0][idx * 4], pg + idx * 4);
  }
  wait_async_lds();
  __syncthreads();

  // persistent per-wave A fragments of the q chunk (unscaled)
  v16h qA[2][4];
  for (int ks = 0; ks < 2; ++ks)
    for (int mt = 0; mt < 4; ++mt)
      qA[ks][mt] = frag2(&sQ[(mt * 16 + ln) * DH + ks * 32 + lh * 8],
                         &sQ[(mt * 16 + ln) * DH + ks * 32 + 16 + lh * 8]);
  float b1e[4];
  for (int et = 0; et < 4; ++et) b1e[et] = sB1[et * 16 + ln];
  const float b2v = b2[0];
  const float inv_sqrt_d = 0.125f;  // 1/sqrt(64)

  // O accumulator tiles: wave w owns (mt = w/2, nt = 2*(w%2) + {0,1})
  const int mtO = wave >> 1;
  const int ntO = (wave & 1) * 2;
  const v8f zero8 = {0.f, 0.f, 0.f, 0.f, 0.f, 0.f, 0.f, 0.f};
  v8f oacc[2];
  oacc[0] = zero8;
  oacc[1] = zero8;

  for (int kb = 0; kb < NC; ++kb) {
    const int cur = kb & 1;

    // transpose this block's V (LDS -> LDS), off the global critical path
    for (int idx = tid; idx < CHUNK * DH; idx += 256) {
      int jc = idx >> 6, dd = idx & 63;
      sVt[dd * CHUNK + jc] = sVraw[cur][idx];
    }

    // prefetch next key block into the other buffer (async, overlapped)
    if (kb + 1 < NC) {
      const int nxt = cur ^ 1;
      const size_t j0n = (size_t)(kb + 1) * CHUNK;
      const half_t* kg = kh + base + j0n * DH;
      const half_t* vg = vh + base + j0n * DH;
      const float*  pg = kp + base + j0n * DH;
      for (int idx = tid; idx < 512; idx += 256) {
        copy16_g2l(&sK[nxt][idx * 8], kg + idx * 8);
        copy16_g2l(&sVraw[nxt][idx * 8], vg + idx * 8);
      }
      for (int idx = tid; idx < 1024; idx += 256)
        copy16_g2l(&sKP[nxt][idx * 4], pg + idx * 4);
    }

    // ---- scores: each wave handles 8 keys of this block ----
    for (int jj = 0; jj < 8; ++jj) {
      const int jc = wave * 8 + jj;
      v16h kv0 = frag2(&sK[cur][jc * DH + lh * 8], &sK[cur][jc * DH + 16 + lh * 8]);
      v16h kv1 = frag2(&sK[cur][jc * DH + 32 + lh * 8],
                       &sK[cur][jc * DH + 48 + lh * 8]);
      float sacc[4][8];
      for (int mt = 0; mt < 4; ++mt)
        for (int r = 0; r < 8; ++r) sacc[mt][r] = 0.0f;

      for (int et = 0; et < 4; ++et) {
        v8f acc[4];
        for (int mt = 0; mt < 4; ++mt) {  // accumulator init = qp fragment
          union { v8f v; v4f q[2]; } u;
          const float* p = &sQPf[(et * 4 + mt) * 256 + lane * 8];
          u.q[0] = *(const v4f*)p;
          u.q[1] = *(const v4f*)(p + 4);
          acc[mt] = u.v;
        }
        for (int ks = 0; ks < 2; ++ks) {
          v16h kv = ks ? kv1 : kv0;
          v16h wB = frag2(&sW1[(et * 16 + ln) * DH + ks * 32 + lh * 8],
                          &sW1[(et * 16 + ln) * DH + ks * 32 + 16 + lh * 8]);
          for (int mt = 0; mt < 4; ++mt) {
            v16h sa = qA[ks][mt] * kv;  // q .* k[j]  (packed f16 mul)
            acc[mt] = wmma_f16(sa, wB, acc[mt]);
          }
        }
        const float kpb = sKP[cur][jc * DH + et * 16 + ln] + b1e[et];
        const float w2e = sW2[et * 16 + ln];
        for (int mt = 0; mt < 4; ++mt)
          for (int r = 0; r < 8; ++r) {
            float hv = acc[mt][r] + kpb;
            sacc[mt][r] += gelu_f(hv) * w2e;
          }
      }
      // reduce over the e dimension (16 lanes) and write scores
      for (int mt = 0; mt < 4; ++mt)
        for (int r = 0; r < 8; ++r) {
          float v = sacc[mt][r];
          v += __shfl_xor(v, 1, 32);
          v += __shfl_xor(v, 2, 32);
          v += __shfl_xor(v, 4, 32);
          v += __shfl_xor(v, 8, 32);
          if (ln == r) {
            int m = mt * 16 + r + 8 * lh;
            sS[m * CHUNK + jc] = (v + b2v) * inv_sqrt_d;
          }
        }
    }
    __syncthreads();

    // ---- online softmax: 4 lanes per query row, all 256 threads ----
    {
      const int row = tid >> 2, sub = tid & 3;
      const float* srow = &sS[row * CHUNK + sub * 16];
      float mold = sMrow[row];
      float bmax = -3.0e38f;
      for (int t = 0; t < 16; ++t) bmax = fmaxf(bmax, srow[t]);
      bmax = fmaxf(bmax, __shfl_xor(bmax, 1, 32));
      bmax = fmaxf(bmax, __shfl_xor(bmax, 2, 32));
      float mnew = fmaxf(mold, bmax);
      float ssum = 0.0f;
      half_t* prow = &sP[row * CHUNK + sub * 16];
      for (int t = 0; t < 16; ++t) {
        float p = fast_exp(srow[t] - mnew);
        ssum += p;
        prow[t] = (half_t)p;
      }
      ssum += __shfl_xor(ssum, 1, 32);
      ssum += __shfl_xor(ssum, 2, 32);
      if (sub == 0) {
        float scale = fast_exp(mold - mnew);
        sLrow[row] = sLrow[row] * scale + ssum;
        sMrow[row] = mnew;
        sScale[row] = scale;
      }
    }
    __syncthreads();

    // ---- O rescale + O += P @ V ----
    {
      v16h pA0 = frag2(&sP[(mtO * 16 + ln) * CHUNK + lh * 8],
                       &sP[(mtO * 16 + ln) * CHUNK + 16 + lh * 8]);
      v16h pA1 = frag2(&sP[(mtO * 16 + ln) * CHUNK + 32 + lh * 8],
                       &sP[(mtO * 16 + ln) * CHUNK + 48 + lh * 8]);
      float sc[8];
      for (int r = 0; r < 8; ++r) sc[r] = sScale[mtO * 16 + r + 8 * lh];
      for (int t = 0; t < 2; ++t) {
        int nt = ntO + t;
        v16h vB0 = frag2(&sVt[(nt * 16 + ln) * CHUNK + lh * 8],
                         &sVt[(nt * 16 + ln) * CHUNK + 16 + lh * 8]);
        v16h vB1 = frag2(&sVt[(nt * 16 + ln) * CHUNK + 32 + lh * 8],
                         &sVt[(nt * 16 + ln) * CHUNK + 48 + lh * 8]);
        for (int r = 0; r < 8; ++r) oacc[t][r] *= sc[r];
        oacc[t] = wmma_f16(pA0, vB0, oacc[t]);
        oacc[t] = wmma_f16(pA1, vB1, oacc[t]);
      }
    }

    wait_async_lds();  // next block's async K/V/KP staging complete
    __syncthreads();
  }

  // finalize: divide by l, store f16 in (B, C, H*d) layout for output GEMM
  for (int t = 0; t < 2; ++t) {
    int nt = ntO + t;
    for (int r = 0; r < 8; ++r) {
      int m = mtO * 16 + r + 8 * lh;
      float o = oacc[t][r] * (1.0f / sLrow[m]);
      oh[((size_t)(b * CC + q0 + m)) * DDIM + hh * DH + nt * 16 + ln] = (half_t)o;
    }
  }
}

// ---------------------------------------------------------------------------
extern "C" void kernel_launch(void* const* d_in, const int* in_sizes, int n_in,
                              void* d_out, int out_size, void* d_ws,
                              size_t ws_size, hipStream_t stream) {
  const float* x  = (const float*)d_in[0];
  const float* Wq = (const float*)d_in[1];
  const float* bq = (const float*)d_in[2];
  const float* Wk = (const float*)d_in[3];
  const float* bk = (const float*)d_in[4];
  const float* Wv = (const float*)d_in[5];
  const float* bv = (const float*)d_in[6];
  const float* w1 = (const float*)d_in[7];
  const float* b1 = (const float*)d_in[8];
  const float* w2 = (const float*)d_in[9];
  const float* b2 = (const float*)d_in[10];
  const float* Wo = (const float*)d_in[11];
  const float* bo = (const float*)d_in[12];
  float* out = (float*)d_out;

  char* ws = (char*)d_ws;
  size_t off = 0;
  auto alloc = [&](size_t bytes) -> void* {
    void* p = ws + off;
    off += (bytes + 255) & ~(size_t)255;
    return p;
  };
  const size_t nX = (size_t)BB * CC * DDIM;      // 1,048,576
  const size_t nW = (size_t)DDIM * DDIM;         // 262,144
  const size_t nHd = (size_t)BB * HH * CC * DH;  // 1,048,576

  half_t* xh    = (half_t*)alloc(nX * 2);
  half_t* wqh   = (half_t*)alloc(nW * 2);
  half_t* wkh   = (half_t*)alloc(nW * 2);
  half_t* wvh   = (half_t*)alloc(nW * 2);
  half_t* woh   = (half_t*)alloc(nW * 2);
  half_t* w1qh  = (half_t*)alloc(4096 * 2);
  half_t* w1kh  = (half_t*)alloc(4096 * 2);
  half_t* w1qkh = (half_t*)alloc(4096 * 2);
  half_t* qhd   = (half_t*)alloc(nHd * 2);
  half_t* khd   = (half_t*)alloc(nHd * 2);
  half_t* vhd   = (half_t*)alloc(nHd * 2);
  float*  qpf   = (float*)alloc(nHd * 4);
  float*  kpf   = (float*)alloc(nHd * 4);
  half_t* ohh   = (half_t*)alloc(nX * 2);

  // f32 -> f16 staging
  f32_to_f16_kernel<<<(int)(nX / 256), 256, 0, stream>>>(x, xh, (int)nX);
  f32_to_f16_kernel<<<(int)(nW / 256), 256, 0, stream>>>(Wq, wqh, (int)nW);
  f32_to_f16_kernel<<<(int)(nW / 256), 256, 0, stream>>>(Wk, wkh, (int)nW);
  f32_to_f16_kernel<<<(int)(nW / 256), 256, 0, stream>>>(Wv, wvh, (int)nW);
  f32_to_f16_kernel<<<(int)(nW / 256), 256, 0, stream>>>(Wo, woh, (int)nW);
  split_w1_kernel<<<16, 256, 0, stream>>>(w1, w1qh, w1kh, w1qkh);

  // Q/K/V projections -> (B,H,C,d) f16   [M=2048 N=512 K=512 -> 512 blocks]
  gemm_wmma_kernel<<<512, 256, 0, stream>>>(xh, wqh, bq, qhd, BB * CC, DDIM,
                                            DDIM, 1);
  gemm_wmma_kernel<<<512, 256, 0, stream>>>(xh, wkh, bk, khd, BB * CC, DDIM,
                                            DDIM, 1);
  gemm_wmma_kernel<<<512, 256, 0, stream>>>(xh, wvh, bv, vhd, BB * CC, DDIM,
                                            DDIM, 1);

  // qp = q @ w1_q^T,  kp = k @ w1_k^T   [M=16384 N=64 K=64 -> 512 blocks]
  gemm_wmma_kernel<<<512, 256, 0, stream>>>(qhd, w1qh, nullptr, qpf,
                                            BB * HH * CC, DH, DH, 0);
  gemm_wmma_kernel<<<512, 256, 0, stream>>>(khd, w1kh, nullptr, kpf,
                                            BB * HH * CC, DH, DH, 0);

  // second-order relational attention: one block per (b,h,chunk)
  sora_attn_kernel<<<BB * HH * NC, 256, 0, stream>>>(qhd, khd, vhd, qpf, kpf,
                                                     w1qkh, b1, w2, b2, ohh);

  // output projection -> f32 d_out
  gemm_wmma_kernel<<<512, 256, 0, stream>>>(ohh, woh, bo, out, BB * CC, DDIM,
                                            DDIM, 0);
}
